// SCN_30949534335323
// MI455X (gfx1250) — matrix operand order repaired and checked by reading, not compiled, not run on hardware
//
#include <hip/hip_runtime.h>

// ---------------------------------------------------------------------------
// Problem dims (fixed by reference)
// ---------------------------------------------------------------------------
#define TT   16
#define NN   32
#define CC   128
#define HH   32
#define WW   32
#define CO   128

typedef __attribute__((ext_vector_type(16))) __bf16         v16bf;
typedef __attribute__((ext_vector_type(8)))  float          v8f;
typedef __attribute__((ext_vector_type(8)))  unsigned short u16x8;

union Frag {
    v16bf bf;
    u16x8 h[2];
};

__device__ __forceinline__ unsigned short f32_to_bf16(float f) {
    unsigned int u = __float_as_uint(f);
    u += 0x7FFFu + ((u >> 16) & 1u);          // round-to-nearest-even
    return (unsigned short)(u >> 16);
}
__device__ __forceinline__ float bf16_to_f32(unsigned short h) {
    return __uint_as_float(((unsigned int)h) << 16);
}
// unpack a channel-pair u32 (two bf16) to floats
__device__ __forceinline__ float pair_lo(unsigned int u) {
    return __uint_as_float(u << 16);
}
__device__ __forceinline__ float pair_hi(unsigned int u) {
    return __uint_as_float(u & 0xFFFF0000u);
}

// ---------------------------------------------------------------------------
// Prep kernel: fold BN scale into pointwise weights, split fp32 -> bf16 hi/lo,
// fold (pw_b - mean)*scale + beta into a single epilogue bias.
// ---------------------------------------------------------------------------
__global__ void snn_prep(const float* __restrict__ pw_w,
                         const float* __restrict__ pw_b,
                         const float* __restrict__ gamma,
                         const float* __restrict__ beta,
                         const float* __restrict__ run_mean,
                         const float* __restrict__ run_var,
                         unsigned short* __restrict__ pw_hi,
                         unsigned short* __restrict__ pw_lo,
                         float* __restrict__ bias2) {
    int i = blockIdx.x * blockDim.x + threadIdx.x;
    if (i >= CO * CC) return;
    int o = i >> 7;
    float scale = gamma[o] * rsqrtf(run_var[o] + 1e-5f);
    float w = pw_w[i] * scale;
    unsigned short hi = f32_to_bf16(w);
    pw_hi[i] = hi;
    pw_lo[i] = f32_to_bf16(w - bf16_to_f32(hi));
    if ((i & 127) == 0)
        bias2[o] = (pw_b[o] - run_mean[o]) * scale + beta[o];
}

// ---------------------------------------------------------------------------
// Fused kernel: LIF scan (state resident in LDS across T) -> depthwise 3x3
// -> bf16x3 split-precision WMMA pointwise GEMM -> folded-BN epilogue.
// Grid: 512 blocks = 32 samples x 16 row-tiles (2 output rows each).
// Block: 256 threads = 8 waves.
// ---------------------------------------------------------------------------
#define ROWS    4      // 2 output rows + 1-row halo each side
#define PIX     64     // 2 rows x 32 cols of output pixels per block
#define SPW     34     // padded spike row width (cols -1..32)
#define CPAD    132    // spike channel stride in halves (bank spread: 2*col)
#define DWS     136    // dwo row stride in halves (bank-conflict-free b128)

__global__ __launch_bounds__(256) void snn_fused(
    const float* __restrict__ x,
    const float* __restrict__ dw_w,
    const float* __restrict__ dw_b,
    const unsigned short* __restrict__ pw_hi,
    const unsigned short* __restrict__ pw_lo,
    const float* __restrict__ bias2,
    float* __restrict__ out) {

    __shared__ float          v_lds [ROWS * CC * WW];     // 65536 B membrane
    __shared__ unsigned short spk   [ROWS * SPW * CPAD];  // 35904 B spikes
    __shared__ unsigned short dwo_hi[PIX * DWS];          // 17408 B
    __shared__ unsigned short dwo_lo[PIX * DWS];          // 17408 B

    const int tid = threadIdx.x;
    const int n  = blockIdx.x >> 4;
    const int rt = blockIdx.x & 15;
    const int r0 = rt * 2;

    // ---- zero-init membrane state and spike tile (includes halo/pad zeros)
    for (int i = tid; i < ROWS * CC * WW;    i += 256) v_lds[i] = 0.0f;
    for (int i = tid; i < ROWS * SPW * CPAD; i += 256) spk[i]   = 0;

    // ---- LIF mapping: lane along W for coalesced x loads
    const int col = tid & 31;
    const int g   = tid >> 5;          // 8 channel groups of 16

    // ---- depthwise mapping: one channel-pair, one row-half per thread
    const int cp  = tid & 63;          // channel pair id, c0 = 2*cp
    const int c0  = cp * 2;
    const int pr  = tid >> 6;          // 0..3
    const int phD = pr >> 1;           // output row 0/1 within tile
    const int hf  = pr & 1;            // column half (0: cols 0-15, 1: 16-31)
    float wA[9], wB[9];
#pragma unroll
    for (int q = 0; q < 9; ++q) {
        wA[q] = dw_w[c0 * 9 + q];
        wB[q] = dw_w[(c0 + 1) * 9 + q];
    }
    const float wbA = dw_b[c0];
    const float wbB = dw_b[c0 + 1];

    // ---- GEMM mapping (per ISA 7.12.2 wave32 layouts)
    const int wv   = tid >> 5;         // wave id -> M strip of 16
    const int lane = tid & 31;
    const int lm   = lane & 15;
    const int hiH  = lane >> 4;        // 0: low half-wave, 1: high
    const int oA   = wv * 16 + lm;     // A-matrix row (output channel)
    float breg[8];
#pragma unroll
    for (int j = 0; j < 8; ++j) breg[j] = bias2[wv * 16 + hiH * 8 + j];

    const size_t tstride = (size_t)NN * CC * HH * WW;
    const unsigned int* sp32 = (const unsigned int*)spk;  // channel-pair view

    __syncthreads();

    for (int t = 0; t < TT; ++t) {
        // =========== Phase 1: LIF step, spikes -> LDS (bf16 0/1) ===========
        for (int row = 0; row < ROWS; ++row) {
            const int ra = r0 - 1 + row;          // absolute image row
            if (ra < 0 || ra >= HH) continue;     // halo outside: stays zero
            const float* xrow =
                x + (((size_t)t * NN + n) * CC * HH + ra) * WW + col;
            const float* xrow_pf = xrow + ((t == TT - 1) ? 0 : tstride);
#pragma unroll
            for (int k = 0; k < 16; k += 2) {
                const int c = g * 16 + k;
                if ((k & 7) == 0)   // keep next-t slab warm, 1 line / 8 chans
                    __builtin_prefetch(xrow_pf + (size_t)c * HH * WW, 0, 0);
                const float x0 = xrow[(size_t)c * HH * WW];
                const float x1 = xrow[(size_t)(c + 1) * HH * WW];
                const int vi = (row * CC + c) * WW + col;
                float v0 = 0.5f * (v_lds[vi] + x0);
                float v1 = 0.5f * (v_lds[vi + WW] + x1);
                const bool s0 = (v0 >= 1.0f);
                const bool s1 = (v1 >= 1.0f);
                v_lds[vi]      = s0 ? 0.0f : v0;
                v_lds[vi + WW] = s1 ? 0.0f : v1;
                const unsigned int pack =
                    (s0 ? 0x3F80u : 0u) | (s1 ? 0x3F800000u : 0u);
                *(unsigned int*)&spk[(row * SPW + col + 1) * CPAD + c] = pack;
            }
        }
        __syncthreads();

        // ====== Phase 2: depthwise 3x3, channel-pair b32 taps, hi/lo split =
        {
            float a0[3], a1[3], b0[3], b1[3];     // rolling window, 2 channels
#pragma unroll
            for (int dr = 0; dr < 3; ++dr) {
                unsigned int u;
                u = sp32[((phD + dr) * SPW + hf * 16 + 0) * (CPAD / 2) + cp];
                a0[dr] = pair_lo(u); b0[dr] = pair_hi(u);
                u = sp32[((phD + dr) * SPW + hf * 16 + 1) * (CPAD / 2) + cp];
                a1[dr] = pair_lo(u); b1[dr] = pair_hi(u);
            }
            for (int jj = 0; jj < 16; ++jj) {
                float accA = wbA, accB = wbB;
#pragma unroll
                for (int dr = 0; dr < 3; ++dr) {
                    const unsigned int u =
                        sp32[((phD + dr) * SPW + hf * 16 + jj + 2) * (CPAD / 2) + cp];
                    const float a2 = pair_lo(u), b2 = pair_hi(u);
                    accA += wA[dr * 3 + 0] * a0[dr] + wA[dr * 3 + 1] * a1[dr]
                          + wA[dr * 3 + 2] * a2;
                    accB += wB[dr * 3 + 0] * b0[dr] + wB[dr * 3 + 1] * b1[dr]
                          + wB[dr * 3 + 2] * b2;
                    a0[dr] = a1[dr]; a1[dr] = a2;
                    b0[dr] = b1[dr]; b1[dr] = b2;
                }
                const int pix = phD * WW + hf * 16 + jj;
                const unsigned int hA = f32_to_bf16(accA);
                const unsigned int hB = f32_to_bf16(accB);
                *(unsigned int*)&dwo_hi[pix * DWS + c0] = hA | (hB << 16);
                const unsigned int lA =
                    f32_to_bf16(accA - bf16_to_f32((unsigned short)hA));
                const unsigned int lB =
                    f32_to_bf16(accB - bf16_to_f32((unsigned short)hB));
                *(unsigned int*)&dwo_lo[pix * DWS + c0] = lA | (lB << 16);
            }
        }
        __syncthreads();

        // ====== Phase 3: pointwise GEMM (M=128,N=64,K=128) via WMMA ======
        // Dual accumulators per wave for WMMA ILP; store offsets fold to
        // immediates off one per-t base pointer.
        float* outp = out +
            ((((size_t)t * NN + n) * CO + (wv * 16 + hiH * 8)) * HH + r0) * WW + lm;
#pragma unroll
        for (int ntp = 0; ntp < 2; ++ntp) {
            v8f acc0 = {0.f, 0.f, 0.f, 0.f, 0.f, 0.f, 0.f, 0.f};
            v8f acc1 = {0.f, 0.f, 0.f, 0.f, 0.f, 0.f, 0.f, 0.f};
            const int pix0 = ntp * 32 + lm;       // row r0+ntp, col lm
            const int pix1 = pix0 + 16;           // row r0+ntp, col lm+16
#pragma unroll
            for (int kb = 0; kb < CC; kb += 32) {
                // A fragment: lane K = kb + hiH*16 + 0..15
                const int ka = kb + hiH * 16;
                Frag Ah, Al, B0h, B0l, B1h, B1l;
                Ah.h[0] = *(const u16x8*)(pw_hi + oA * CC + ka);
                Ah.h[1] = *(const u16x8*)(pw_hi + oA * CC + ka + 8);
                Al.h[0] = *(const u16x8*)(pw_lo + oA * CC + ka);
                Al.h[1] = *(const u16x8*)(pw_lo + oA * CC + ka + 8);
                // B fragments: lane = pixel (N), K = kb + hiH*16 + 0..15
                const int kc = kb + hiH * 16;
                B0h.h[0] = *(const u16x8*)(dwo_hi + pix0 * DWS + kc);
                B0h.h[1] = *(const u16x8*)(dwo_hi + pix0 * DWS + kc + 8);
                B0l.h[0] = *(const u16x8*)(dwo_lo + pix0 * DWS + kc);
                B0l.h[1] = *(const u16x8*)(dwo_lo + pix0 * DWS + kc + 8);
                B1h.h[0] = *(const u16x8*)(dwo_hi + pix1 * DWS + kc);
                B1h.h[1] = *(const u16x8*)(dwo_hi + pix1 * DWS + kc + 8);
                B1l.h[0] = *(const u16x8*)(dwo_lo + pix1 * DWS + kc);
                B1l.h[1] = *(const u16x8*)(dwo_lo + pix1 * DWS + kc + 8);
                // 3-term split precision, two independent accumulators
                acc0 = __builtin_amdgcn_wmma_f32_16x16x32_bf16(
                           false, Ah.bf, false, B0h.bf, (short)0, acc0, false, false);
                acc1 = __builtin_amdgcn_wmma_f32_16x16x32_bf16(
                           false, Ah.bf, false, B1h.bf, (short)0, acc1, false, false);
                acc0 = __builtin_amdgcn_wmma_f32_16x16x32_bf16(
                           false, Ah.bf, false, B0l.bf, (short)0, acc0, false, false);
                acc1 = __builtin_amdgcn_wmma_f32_16x16x32_bf16(
                           false, Ah.bf, false, B1l.bf, (short)0, acc1, false, false);
                acc0 = __builtin_amdgcn_wmma_f32_16x16x32_bf16(
                           false, Al.bf, false, B0h.bf, (short)0, acc0, false, false);
                acc1 = __builtin_amdgcn_wmma_f32_16x16x32_bf16(
                           false, Al.bf, false, B1h.bf, (short)0, acc1, false, false);
            }
            // D layout: VGPR j -> M = wv*16 + hiH*8 + j
#pragma unroll
            for (int j = 0; j < 8; ++j) {
                outp[(size_t)j * HH * WW + ntp * WW]      = acc0[j] + breg[j];
                outp[(size_t)j * HH * WW + ntp * WW + 16] = acc1[j] + breg[j];
            }
        }
        // No barrier needed here: next phase-1 writes spk (consumed before the
        // phase-2 barrier), and next phase-2's dwo writes are fenced by the
        // barrier after phase 1.
    }
}

// ---------------------------------------------------------------------------
// Launch
// ---------------------------------------------------------------------------
extern "C" void kernel_launch(void* const* d_in, const int* in_sizes, int n_in,
                              void* d_out, int out_size, void* d_ws, size_t ws_size,
                              hipStream_t stream) {
    const float* x        = (const float*)d_in[0];
    const float* dw_w     = (const float*)d_in[1];
    const float* dw_b     = (const float*)d_in[2];
    const float* pw_w     = (const float*)d_in[3];
    const float* pw_b     = (const float*)d_in[4];
    const float* gamma    = (const float*)d_in[5];
    const float* beta     = (const float*)d_in[6];
    const float* run_mean = (const float*)d_in[7];
    const float* run_var  = (const float*)d_in[8];
    float* out = (float*)d_out;

    unsigned short* pw_hi = (unsigned short*)d_ws;
    unsigned short* pw_lo = pw_hi + CO * CC;
    float*          bias2 = (float*)(pw_lo + CO * CC);

    snn_prep<<<(CO * CC + 255) / 256, 256, 0, stream>>>(
        pw_w, pw_b, gamma, beta, run_mean, run_var, pw_hi, pw_lo, bias2);

    snn_fused<<<NN * (HH / 2), 256, 0, stream>>>(
        x, dw_w, dw_b, pw_hi, pw_lo, bias2, out);
}
